// SegmentationAttentionSeparateModule_31954556682489
// MI455X (gfx1250) — compile-verified, matrix operand order checked by reading, not run on hardware
//
#include <hip/hip_runtime.h>
#include <hip/hip_bf16.h>

// Flash-attention style maskRead for MI455X (gfx1250, wave32, WMMA bf16).
// qkey [B,Dk,Q] f32, mkey [B,Dk,M] f32, mval [B,Dv,M] f32, masks bool(u8).
// out [B,Dv,Q] f32.  B=4 Dk=128 Dv=512 Q=4096 M=8192, P_SCALAR=40.
//
// Roofline: ~172 GFLOP vs ~120MB HBM -> compute bound; use
// v_wmma_f32_16x16x32_bf16 (f32 accum), single pass over M with online
// softmax, per-batch V (16MB) stays L2-resident across query-tile blocks.

typedef __attribute__((ext_vector_type(16))) __bf16 v16bf;
typedef __attribute__((ext_vector_type(8)))  __bf16 v8bf;
typedef __attribute__((ext_vector_type(8)))  float  v8f;

#define BATCH   4
#define DK      128
#define DV      512
#define QTOT    4096
#define MTOT    8192
#define QT      64      // query tile per workgroup
#define MC      64      // memory chunk per iteration
#define NCHUNK  (MTOT / MC)
#define PSCALAR 40.0f

// padded LDS row strides (keep 16B alignment of 128b ds loads: strides are x16 elems? bytes: 136*2=272=16*17 ok, 72*2=144=16*9 ok)
#define KROW 136   // bf16 elems per row of K/Q stage (128 + pad)
#define SROW 65    // f32 elems per row of score buffer
#define PROW 72    // bf16 elems per row of P buffer (64 + pad)

__device__ __forceinline__ v16bf frag_from_lds(const __bf16* row, int ko, int h) {
  // ISA 16-bit A/B fragment: lanes 0-15 -> K = ko+[0..7] and ko+16+[0..7],
  // lanes 16-31 same rows/cols with K offset +8 (h==8).
  union { v16bf v; v8bf s[2]; } u;
  u.s[0] = *(const v8bf*)(row + ko + h);        // ds_load_b128
  u.s[1] = *(const v8bf*)(row + ko + 16 + h);   // ds_load_b128
  return u.v;
}

__device__ __forceinline__ v16bf frag_from_global_f32(const float* p) {
  // p -> first 8 consecutive f32 (K=ko+h..), second segment at +16 elems.
  v8f a = *(const v8f*)(p);          // 2x global_load_b128
  v8f c = *(const v8f*)(p + 16);     // 2x global_load_b128
  v16bf r;
#pragma unroll
  for (int i = 0; i < 8; ++i) { r[i] = (__bf16)a[i]; r[8 + i] = (__bf16)c[i]; }
  return r;
}

__global__ void __launch_bounds__(256)
mask_read_attn_kernel(const float* __restrict__ qkey,
                      const float* __restrict__ mkey,
                      const float* __restrict__ mval,
                      const unsigned char* __restrict__ qmask,
                      const unsigned char* __restrict__ mmask,
                      float* __restrict__ out) {
  __shared__ __bf16 lds_q[QT * KROW];        // [q][d] bf16
  __shared__ __bf16 lds_k[MC * KROW];        // [m][d] bf16
  __shared__ float  lds_s[MC * SROW];        // [m][q] f32 scores
  __shared__ __bf16 lds_p[QT * PROW];        // [q][m] bf16 probabilities
  __shared__ float  lds_red[4 * QT];
  __shared__ float  row_max[QT];
  __shared__ float  row_sum[QT];
  __shared__ float  lds_alpha[QT];
  __shared__ unsigned char lds_mm[MC];

  const int tid  = threadIdx.x;
  const int wv   = tid >> 5;          // wave id 0..7
  const int lane = tid & 31;
  const int row  = lane & 15;         // tile row/col within fragment
  const int hb   = (lane >> 4) << 3;  // K sub-offset 0 or 8
  const int b    = blockIdx.x >> 6;
  const int q0   = (blockIdx.x & 63) * QT;
  const int dv0  = wv * 64;           // this wave's 64 Dv rows

  // ---- stage Q tile: qkey[b, d, q0+ql] -> lds_q[ql][d] (bf16) ----
  for (int idx = tid; idx < DK * QT; idx += 256) {
    int d = idx >> 6, ql = idx & 63;
    float v = qkey[((size_t)b * DK + d) * QTOT + q0 + ql];
    lds_q[ql * KROW + d] = (__bf16)v;
  }
  if (tid < QT) { row_max[tid] = -3.0e38f; row_sum[tid] = 0.0f; }

  v8f acc[4][4];   // [dv tile][q tile] 16x16 f32 accumulators
#pragma unroll
  for (int dt = 0; dt < 4; ++dt)
#pragma unroll
    for (int qt = 0; qt < 4; ++qt) acc[dt][qt] = v8f{};

  __syncthreads();

  for (int ch = 0; ch < NCHUNK; ++ch) {
    const int m0 = ch * MC;

    // ---- stage K chunk: mkey[b, d, m0+ml] -> lds_k[ml][d]; mask bytes ----
    for (int idx = tid; idx < DK * MC; idx += 256) {
      int d = idx >> 6, ml = idx & 63;
      float v = mkey[((size_t)b * DK + d) * MTOT + m0 + ml];
      lds_k[ml * KROW + d] = (__bf16)v;
    }
    if (tid < MC) lds_mm[tid] = mmask[(size_t)b * MTOT + m0 + tid];
    __syncthreads();

    // ---- S = K^T Q : 16 tiles of 16x16, 2 per wave, 4 K-steps (K=32) ----
#pragma unroll
    for (int tt = 0; tt < 2; ++tt) {
      const int t  = wv * 2 + tt;
      const int mi = t >> 2, qi = t & 3;
      const __bf16* arow = &lds_k[(mi * 16 + row) * KROW];
      const __bf16* brow = &lds_q[(qi * 16 + row) * KROW];
      v8f s = v8f{};
#pragma unroll
      for (int ks = 0; ks < 4; ++ks) {
        v16bf a  = frag_from_lds(arow, ks * 32, hb);
        v16bf bq = frag_from_lds(brow, ks * 32, hb);
        s = __builtin_amdgcn_wmma_f32_16x16x32_bf16(false, a, false, bq,
                                                    (short)0, s, false, false);
      }
      const int qcol = qi * 16 + row;
      const int mb   = mi * 16 + hb;
#pragma unroll
      for (int r = 0; r < 8; ++r)
        lds_s[(mb + r) * SROW + qcol] = s[r] * PSCALAR;
    }
    __syncthreads();

    // ---- online softmax over this chunk's 64 m-rows (per query column) ----
    {
      const int c = tid & 63, part = tid >> 6;   // 4 thread-groups of 16 rows
      float mx = -3.0e38f;
#pragma unroll
      for (int r = 0; r < 16; ++r) {
        int m = part * 16 + r;
        float s = lds_mm[m] ? lds_s[m * SROW + c] : -1.0e30f;
        mx = fmaxf(mx, s);
      }
      lds_red[part * 64 + c] = mx;
      __syncthreads();
      if (part == 0) {
        float m4 = fmaxf(fmaxf(lds_red[c], lds_red[64 + c]),
                         fmaxf(lds_red[128 + c], lds_red[192 + c]));
        float oldm = row_max[c];
        float newm = fmaxf(oldm, m4);
        row_max[c] = newm;
        lds_alpha[c] = __expf(oldm - newm);
      }
      __syncthreads();
      float newm = row_max[c];
      float ps = 0.0f;
#pragma unroll
      for (int r = 0; r < 16; ++r) {
        int m = part * 16 + r;
        float s = lds_mm[m] ? lds_s[m * SROW + c] : -1.0e30f;
        float p = __expf(s - newm);
        lds_p[c * PROW + m] = (__bf16)p;
        ps += p;
      }
      lds_red[part * 64 + c] = ps;
      __syncthreads();
      if (part == 0) {
        float s4 = lds_red[c] + lds_red[64 + c] + lds_red[128 + c] + lds_red[192 + c];
        row_sum[c] = row_sum[c] * lds_alpha[c] + s4;
      }
      __syncthreads();
    }

    // ---- O = alpha*O + V P : each wave 64 Dv rows x 64 q, 2 K-steps ----
#pragma unroll
    for (int qt = 0; qt < 4; ++qt) {
      float al = lds_alpha[qt * 16 + row];
#pragma unroll
      for (int dt = 0; dt < 4; ++dt)
#pragma unroll
        for (int i = 0; i < 8; ++i) acc[dt][qt][i] *= al;
    }
#pragma unroll
    for (int ks = 0; ks < 2; ++ks) {
      const int ko = ks * 32;
      v16bf va[4];
#pragma unroll
      for (int dt = 0; dt < 4; ++dt) {
        const float* vp = mval + ((size_t)b * DV + dv0 + dt * 16 + row) * (size_t)MTOT
                               + m0 + ko + hb;
        va[dt] = frag_from_global_f32(vp);
      }
#pragma unroll
      for (int qt = 0; qt < 4; ++qt) {
        v16bf pb = frag_from_lds(&lds_p[(qt * 16 + row) * PROW], ko, hb);
#pragma unroll
        for (int dt = 0; dt < 4; ++dt)
          acc[dt][qt] = __builtin_amdgcn_wmma_f32_16x16x32_bf16(false, va[dt], false, pb,
                                                                (short)0, acc[dt][qt],
                                                                false, false);
      }
    }
    // no trailing sync needed: next stage only rewrites lds_k/lds_mm which PV
    // does not read, and the stage->S sync orders everything after that.
    __syncthreads();
  }

  // ---- epilogue: O / l, query mask, store ----
#pragma unroll
  for (int qt = 0; qt < 4; ++qt) {
    const int q = q0 + qt * 16 + row;
    const float inv = 1.0f / row_sum[qt * 16 + row];
    const float sc  = qmask[(size_t)b * QTOT + q] ? inv : 0.0f;
#pragma unroll
    for (int dt = 0; dt < 4; ++dt) {
      const int dvr = dv0 + dt * 16 + hb;
#pragma unroll
      for (int r = 0; r < 8; ++r)
        out[((size_t)b * DV + dvr + r) * QTOT + q] = acc[dt][qt][r] * sc;
    }
  }
}

extern "C" void kernel_launch(void* const* d_in, const int* in_sizes, int n_in,
                              void* d_out, int out_size, void* d_ws, size_t ws_size,
                              hipStream_t stream) {
  (void)in_sizes; (void)n_in; (void)d_ws; (void)ws_size; (void)out_size;
  const float* qkey = (const float*)d_in[0];
  const float* mkey = (const float*)d_in[1];
  const float* mval = (const float*)d_in[2];
  const unsigned char* qmask = (const unsigned char*)d_in[3];
  const unsigned char* mmask = (const unsigned char*)d_in[4];
  float* out = (float*)d_out;

  dim3 grid(BATCH * (QTOT / QT));   // 256 workgroups
  dim3 block(256);                  // 8 wave32 waves
  mask_read_attn_kernel<<<grid, block, 0, stream>>>(qkey, mkey, mval, qmask, mmask, out);
}
